// rits_16174846837120
// MI455X (gfx1250) — matrix operand order repaired and checked by reading, not compiled, not run on hardware
//
#include <hip/hip_runtime.h>

// ---------------------------------------------------------------------------
// RITS forward (BRITS-style) for gfx1250.
// B=512, T=256, F=128, H=512.  bf16 WMMA (16x16x32) GEMMs, f32 accumulate.
// Weights pre-packed to bf16 W^T once per launch -> B frag = 16 strided global u16 loads.
// Activations staged in LDS as bf16              -> A frag = 2x ds b128.
// Input tiles staged via GLOBAL_LOAD_ASYNC_TO_LDS (ASYNCcnt) when available.
// ---------------------------------------------------------------------------

typedef __attribute__((ext_vector_type(16))) __bf16 v16bf;
typedef __attribute__((ext_vector_type(8)))  float  v8f;
typedef __attribute__((ext_vector_type(4)))  int    v4i;

#define AS1 __attribute__((address_space(1)))
#define AS3 __attribute__((address_space(3)))

#if defined(__gfx1250__) && __has_builtin(__builtin_amdgcn_global_load_async_to_lds_b128) && \
    __has_builtin(__builtin_amdgcn_s_wait_asynccnt)
#define USE_ASYNC 1
#else
#define USE_ASYNC 0
#endif

static constexpr int Bsz  = 512;
static constexpr int Tlen = 256;
static constexpr int Fd   = 128;
static constexpr int Hd   = 512;

// bf16 LDS strides (halves): multiple of 8 (16B align), dword-stride % 64 == 4
static constexpr int LDFh = 136;   // 128-wide bf16 tiles
static constexpr int LDCh = 264;   // 256-wide bf16 tiles
static constexpr int LDHh = 520;   // 512-wide bf16 tiles
// f32 LDS stride: 132 floats = 528B (16B multiple, dword-stride % 64 == 4)
static constexpr int LDF  = 132;

__device__ __forceinline__ unsigned short f2bf(float f) {
  union { float f; unsigned u; } v; v.f = f;
  unsigned r = v.u + 0x7FFFu + ((v.u >> 16) & 1u);   // round-to-nearest-even
  return (unsigned short)(r >> 16);
}

#if USE_ASYNC
__device__ __forceinline__ void async_b128(const float* g, float* l) {
  __builtin_amdgcn_global_load_async_to_lds_b128(
      (AS1 v4i*)(void*)(const void*)g, (AS3 v4i*)(void*)l, 0, 0);
}
#endif

// A fragment from bf16 LDS tile (row m contiguous in k):
// lane = M, kb = k0 + 8*(lane>=16); elements = k in [kb,kb+8) U [kb+16,kb+24)
__device__ __forceinline__ v16bf load_ab(const unsigned short* s, int LD, int k0, int lane) {
  int m  = lane & 15;
  int kb = k0 + ((lane >> 4) & 1) * 8;
  const unsigned short* p = s + m * LD + kb;
  union { v16bf v; uint4 q[2]; } r;
  r.q[0] = *(const uint4*)p;
  r.q[1] = *(const uint4*)(p + 16);
  return r.v;
}

// B fragment from pre-packed bf16 W^T (Wt[k][n], n contiguous):
// lane = N (n0+lane%16), kb = k0 + 8*(lane>=16); elements = k in [kb,kb+8) U [kb+16,kb+24)
__device__ __forceinline__ v16bf load_bp(const unsigned short* __restrict__ Wt,
                                         int N, int n0, int k0, int lane) {
  const unsigned short* p = Wt + (size_t)(k0 + ((lane >> 4) & 1) * 8) * N + n0 + (lane & 15);
  union { v16bf v; unsigned short u[16]; } r;
#pragma unroll
  for (int s = 0; s < 16; ++s) r.u[s] = p[(size_t)(s + (s & 8)) * N];
  return r.v;
}

__device__ __forceinline__ v8f wmma_bf16(v16bf a, v16bf b, v8f c) {
  return __builtin_amdgcn_wmma_f32_16x16x32_bf16(false, a, false, b,
                                                 (short)0, c, false, false);
}

__device__ __forceinline__ float sigmoidf_(float x) {
  return 1.f / (1.f + __expf(-x));
}

#define V8ZERO {0.f,0.f,0.f,0.f,0.f,0.f,0.f,0.f}

// ---------------------------------------------------------------------------

__global__ void rits_init(float* ws, int n) {
  int i = blockIdx.x * blockDim.x + threadIdx.x;
  if (i < n) ws[i] = 0.f;
}

// Wt[k*N + n] = bf16(W[n*K + k])
__global__ void rits_pack(const float* __restrict__ W, unsigned short* __restrict__ Wt,
                          int N, int K) {
  int i = blockIdx.x * blockDim.x + threadIdx.x;
  if (i < N * K) {
    int k = i / N, n = i - k * N;
    Wt[i] = f2bf(W[(size_t)n * K + k]);
  }
}

__global__ __launch_bounds__(256) void rits_step(
    const float* __restrict__ data,
    const unsigned short* __restrict__ Wt_dh,   const float* __restrict__ b_dh,
    const float* __restrict__ w_dx,             const float* __restrict__ b_dx,
    const unsigned short* __restrict__ Wt_hist, const float* __restrict__ b_hist,
    const unsigned short* __restrict__ Wt_feat, const float* __restrict__ b_feat,
    const float* __restrict__ W_feat,
    const unsigned short* __restrict__ Wt_comb, const float* __restrict__ b_comb,
    const unsigned short* __restrict__ Wt_ih,   const unsigned short* __restrict__ Wt_hh,
    const float* __restrict__ b_ih,             const float* __restrict__ b_hh,
    float* __restrict__ hstate, float* __restrict__ cstate,
    float* __restrict__ lossacc, float* __restrict__ imps, int t) {

  // f32 tiles (elementwise math + async staging targets)
  __shared__ __align__(16) float s_d [16 * LDF];
  __shared__ __align__(16) float s_x [16 * LDF];
  __shared__ __align__(16) float s_m [16 * LDF];
  __shared__ __align__(16) float s_s1[16 * LDF];
  __shared__ __align__(16) float s_s2[16 * LDF];
  __shared__ float s_xh[16 * LDF];
  __shared__ float s_xc[16 * LDF];
  __shared__ float s_zh[16 * LDF];
  __shared__ float s_al[16 * LDF];
  // bf16 tiles feeding WMMA A operands
  __shared__ __align__(16) unsigned short s_db  [16 * LDFh];  // deltas
  __shared__ __align__(16) unsigned short s_xcb [16 * LDFh];  // x_c
  __shared__ __align__(16) unsigned short s_catb[16 * LDCh];  // [gamma_x | m]
  __shared__ __align__(16) unsigned short s_hb  [16 * LDHh];  // decayed h
  __shared__ __align__(16) unsigned short s_inpb[16 * LDHh];  // [c_c|m|s1|s2]
  // staged constants
  __shared__ float s_bdh[Hd], s_bg[4 * Hd];
  __shared__ float s_bhist[Fd], s_bfeat[Fd], s_bcomb[Fd], s_diag[Fd];
  __shared__ float s_wdx[Fd], s_bdx[Fd];
  __shared__ float s_red[4];

  const int tid  = threadIdx.x;
  const int wave = tid >> 5;
  const int lane = tid & 31;
  const int row0 = blockIdx.x * 16;
  const size_t slice = (size_t)Tlen * Fd;

  // ---- stage time-t tiles: 512 16-byte chunks per tile, 2 chunks/thread ----
  {
#pragma unroll
    for (int i = 0; i < 2; ++i) {
      int chunk = tid + i * 256;          // 0..511
      int r  = chunk >> 5;                // 32 chunks of 4 floats per row
      int c4 = (chunk & 31) * 4;
      size_t o = (((size_t)(row0 + r) * 10) * Tlen + t) * Fd + c4;
      int l = r * LDF + c4;
#if USE_ASYNC
      async_b128(data + o,             &s_x [l]);
      async_b128(data + o + slice,     &s_m [l]);
      async_b128(data + o + 2 * slice, &s_d [l]);
      async_b128(data + o + 3 * slice, &s_s1[l]);
      async_b128(data + o + 4 * slice, &s_s2[l]);
#else
      *(float4*)&s_x [l] = *(const float4*)(data + o);
      *(float4*)&s_m [l] = *(const float4*)(data + o + slice);
      *(float4*)&s_d [l] = *(const float4*)(data + o + 2 * slice);
      *(float4*)&s_s1[l] = *(const float4*)(data + o + 3 * slice);
      *(float4*)&s_s2[l] = *(const float4*)(data + o + 4 * slice);
#endif
      if (t + 1 < Tlen) {                 // warm L2 for next timestep's tiles
        __builtin_prefetch(data + o + Fd, 0, 1);
        __builtin_prefetch(data + o + slice + Fd, 0, 1);
        __builtin_prefetch(data + o + 2 * slice + Fd, 0, 1);
        __builtin_prefetch(data + o + 3 * slice + Fd, 0, 1);
        __builtin_prefetch(data + o + 4 * slice + Fd, 0, 1);
      }
    }
  }

  // ---- stage biases / per-feature constants (overlaps async copies) ----
  for (int j = tid; j < Hd; j += 256) s_bdh[j] = b_dh[j];
  for (int j = tid; j < 4 * Hd; j += 256) s_bg[j] = b_ih[j] + b_hh[j];
  if (tid < Fd) {
    s_bhist[tid] = b_hist[tid];
    s_bfeat[tid] = b_feat[tid];
    s_bcomb[tid] = b_comb[tid];
    s_diag[tid]  = W_feat[(size_t)tid * Fd + tid];
    s_wdx[tid]   = w_dx[tid];
    s_bdx[tid]   = b_dx[tid];
  }
  if (tid < 4) s_red[tid] = 0.f;

#if USE_ASYNC
  __builtin_amdgcn_s_wait_asynccnt(0);
#endif
  __syncthreads();

  // ---- bf16 conversion pass: d, m, s1, s2 -> WMMA A-operand buffers ----
#pragma unroll
  for (int i = 0; i < 8; ++i) {
    int e = tid + i * 256;
    int r = e >> 7, col = e & 127;
    s_db  [r * LDFh + col]       = f2bf(s_d [r * LDF + col]);
    s_inpb[r * LDHh + 128 + col] = f2bf(s_m [r * LDF + col]);
    s_inpb[r * LDHh + 256 + col] = f2bf(s_s1[r * LDF + col]);
    s_inpb[r * LDHh + 384 + col] = f2bf(s_s2[r * LDF + col]);
  }
  __syncthreads();

  // ---- stage 1: gamma_h = exp(-relu(d @ W_dh^T + b)); h = h * gamma_h ----
  for (int tile = wave; tile < Hd / 16; tile += 8) {
    int n0 = tile * 16;
    v8f acc = V8ZERO;
#pragma unroll
    for (int k0 = 0; k0 < Fd; k0 += 32)
      acc = wmma_bf16(load_ab(s_db, LDFh, k0, lane),
                      load_bp(Wt_dh, Hd, n0, k0, lane), acc);
#pragma unroll
    for (int r = 0; r < 8; ++r) {
      int M = r + ((lane >> 4) << 3);
      int N = n0 + (lane & 15);
      float g = __expf(-fmaxf(acc[r] + s_bdh[N], 0.f));
      s_hb[M * LDHh + N] = f2bf(hstate[(size_t)(row0 + M) * Hd + N] * g);
    }
  }
  __syncthreads();

  // ---- stage 2: x_h = h @ W_hist^T + b_hist ----
  {
    int n0 = wave * 16;
    v8f acc = V8ZERO;
#pragma unroll 4
    for (int k0 = 0; k0 < Hd; k0 += 32)
      acc = wmma_bf16(load_ab(s_hb, LDHh, k0, lane),
                      load_bp(Wt_hist, Fd, n0, k0, lane), acc);
#pragma unroll
    for (int r = 0; r < 8; ++r) {
      int M = r + ((lane >> 4) << 3);
      int N = n0 + (lane & 15);
      s_xh[M * LDF + N] = acc[r] + s_bhist[N];
    }
  }
  __syncthreads();

  // ---- elementwise A: loss1, x_c, gamma_x ----
  float n1 = 0.f, n2 = 0.f, n3 = 0.f, den = 0.f;
#pragma unroll
  for (int i = 0; i < 8; ++i) {
    int e = tid + i * 256;
    int r = e >> 7, col = e & 127;
    float x  = s_x[r * LDF + col];
    float m  = s_m[r * LDF + col];
    float xh = s_xh[r * LDF + col];
    float d  = s_d[r * LDF + col];
    n1 += fabsf(x - xh) * m;  den += m;
    float xc = m * x + (1.f - m) * xh;
    s_xc [r * LDF  + col] = xc;
    s_xcb[r * LDFh + col] = f2bf(xc);
    float gx = __expf(-fmaxf(d * s_wdx[col] + s_bdx[col], 0.f));
    s_catb[r * LDCh + col]       = f2bf(gx);
    s_catb[r * LDCh + 128 + col] = f2bf(m);
  }
  __syncthreads();

  // ---- stages 3+4: z_h = x_c @ Wf^T + b (diag removed); alpha = cat @ Wc^T + b ----
  {
    int n0 = wave * 16;
    v8f az = V8ZERO, aa = V8ZERO;
#pragma unroll
    for (int k0 = 0; k0 < Fd; k0 += 32)
      az = wmma_bf16(load_ab(s_xcb, LDFh, k0, lane),
                     load_bp(Wt_feat, Fd, n0, k0, lane), az);
#pragma unroll
    for (int k0 = 0; k0 < 2 * Fd; k0 += 32)
      aa = wmma_bf16(load_ab(s_catb, LDCh, k0, lane),
                     load_bp(Wt_comb, Fd, n0, k0, lane), aa);
#pragma unroll
    for (int r = 0; r < 8; ++r) {
      int M = r + ((lane >> 4) << 3);
      int N = n0 + (lane & 15);
      s_zh[M * LDF + N] = az[r] + s_bfeat[N] - s_diag[N] * s_xc[M * LDF + N];
      s_al[M * LDF + N] = aa[r] + s_bcomb[N];
    }
  }
  __syncthreads();

  // ---- elementwise B: loss2, c_h, loss3, c_c; emit imputation ----
#pragma unroll
  for (int i = 0; i < 8; ++i) {
    int e = tid + i * 256;
    int r = e >> 7, col = e & 127;
    float x  = s_x[r * LDF + col];
    float m  = s_m[r * LDF + col];
    float xh = s_xh[r * LDF + col];
    float zh = s_zh[r * LDF + col];
    float al = s_al[r * LDF + col];
    n2 += fabsf(x - zh) * m;
    float ch = al * zh + (1.f - al) * xh;
    n3 += fabsf(x - ch) * m;
    float cc = m * x + (1.f - m) * ch;
    s_inpb[r * LDHh + col] = f2bf(cc);
    imps[((size_t)(row0 + r) * Tlen + t) * Fd + col] = cc;
  }
  atomicAdd(&s_red[0], n1);
  atomicAdd(&s_red[1], n2);
  atomicAdd(&s_red[2], n3);
  atomicAdd(&s_red[3], den);
  __syncthreads();
  if (tid < 4) atomicAdd(&lossacc[4 * t + tid], s_red[tid]);

  // ---- stage 5: gates = inp @ W_ih^T + h @ W_hh^T; LSTM update ----
  for (int ht = wave; ht < Hd / 16; ht += 8) {
    int j0 = ht * 16;
    v8f ai = V8ZERO, af = V8ZERO, ag = V8ZERO, ao = V8ZERO;
#pragma unroll 2
    for (int k0 = 0; k0 < 4 * Fd; k0 += 32) {
      if (k0 + 64 < 4 * Fd)   // warm L0/L2 for upcoming B rows
        __builtin_prefetch(Wt_ih + (size_t)(k0 + 64 + lane) * (4 * Hd) + j0, 0, 0);
      v16bf av = load_ab(s_inpb, LDHh, k0, lane);
      ai = wmma_bf16(av, load_bp(Wt_ih, 4 * Hd, j0,          k0, lane), ai);
      af = wmma_bf16(av, load_bp(Wt_ih, 4 * Hd, Hd + j0,     k0, lane), af);
      ag = wmma_bf16(av, load_bp(Wt_ih, 4 * Hd, 2 * Hd + j0, k0, lane), ag);
      ao = wmma_bf16(av, load_bp(Wt_ih, 4 * Hd, 3 * Hd + j0, k0, lane), ao);
      v16bf hv = load_ab(s_hb, LDHh, k0, lane);
      ai = wmma_bf16(hv, load_bp(Wt_hh, 4 * Hd, j0,          k0, lane), ai);
      af = wmma_bf16(hv, load_bp(Wt_hh, 4 * Hd, Hd + j0,     k0, lane), af);
      ag = wmma_bf16(hv, load_bp(Wt_hh, 4 * Hd, 2 * Hd + j0, k0, lane), ag);
      ao = wmma_bf16(hv, load_bp(Wt_hh, 4 * Hd, 3 * Hd + j0, k0, lane), ao);
    }
#pragma unroll
    for (int r = 0; r < 8; ++r) {
      int M = r + ((lane >> 4) << 3);
      int j = j0 + (lane & 15);
      size_t idx = (size_t)(row0 + M) * Hd + j;
      float gi = sigmoidf_(ai[r] + s_bg[j]);
      float gf = sigmoidf_(af[r] + s_bg[Hd + j]);
      float gg = tanhf    (ag[r] + s_bg[2 * Hd + j]);
      float go = sigmoidf_(ao[r] + s_bg[3 * Hd + j]);
      float cn = gf * cstate[idx] + gi * gg;
      cstate[idx] = cn;
      hstate[idx] = go * tanhf(cn);
    }
  }
}

__global__ void rits_fin(const float* __restrict__ lossacc, float* __restrict__ out) {
  float l = 0.f;
  for (int t = 0; t < Tlen; ++t) {
    float d = lossacc[4 * t + 3] + 1e-5f;
    l += (lossacc[4 * t] + lossacc[4 * t + 1] + lossacc[4 * t + 2]) / d;
  }
  out[0] = l;
}

// ---------------------------------------------------------------------------

extern "C" void kernel_launch(void* const* d_in, const int* in_sizes, int n_in,
                              void* d_out, int out_size, void* d_ws, size_t ws_size,
                              hipStream_t stream) {
  const float* data   = (const float*)d_in[0];
  const float* W_dh   = (const float*)d_in[1];
  const float* b_dh   = (const float*)d_in[2];
  const float* w_dx   = (const float*)d_in[3];
  const float* b_dx   = (const float*)d_in[4];
  const float* W_hist = (const float*)d_in[5];
  const float* b_hist = (const float*)d_in[6];
  const float* W_feat = (const float*)d_in[7];
  const float* b_feat = (const float*)d_in[8];
  const float* W_comb = (const float*)d_in[9];
  const float* b_comb = (const float*)d_in[10];
  const float* W_ih   = (const float*)d_in[11];
  const float* W_hh   = (const float*)d_in[12];
  const float* b_ih   = (const float*)d_in[13];
  const float* b_hh   = (const float*)d_in[14];

  float* out     = (float*)d_out;
  float* hstate  = (float*)d_ws;
  float* cstate  = hstate + (size_t)Bsz * Hd;
  float* lossacc = cstate + (size_t)Bsz * Hd;
  float* imps    = out + 1;   // output tuple: (x_loss scalar, imputations[B,T,F])

  // packed bf16 W^T region (16B aligned: preceding f32 region is 16B multiple)
  unsigned short* wp      = (unsigned short*)(lossacc + 4 * Tlen);
  unsigned short* wt_dh   = wp;                                    // 128 x 512
  unsigned short* wt_hist = wt_dh   + (size_t)Fd * Hd;             // 512 x 128
  unsigned short* wt_feat = wt_hist + (size_t)Hd * Fd;             // 128 x 128
  unsigned short* wt_comb = wt_feat + (size_t)Fd * Fd;             // 256 x 128
  unsigned short* wt_ih   = wt_comb + (size_t)2 * Fd * Fd;         // 512 x 2048
  unsigned short* wt_hh   = wt_ih   + (size_t)(4 * Fd) * (4 * Hd); // 512 x 2048

  int n_init = 2 * Bsz * Hd + 4 * Tlen;
  rits_init<<<(n_init + 255) / 256, 256, 0, stream>>>((float*)d_ws, n_init);

  auto pk = [&](const float* W, unsigned short* Wt, int N, int K) {
    rits_pack<<<(N * K + 255) / 256, 256, 0, stream>>>(W, Wt, N, K);
  };
  pk(W_dh,   wt_dh,   Hd,     Fd);      // K=128, N=512
  pk(W_hist, wt_hist, Fd,     Hd);      // K=512, N=128
  pk(W_feat, wt_feat, Fd,     Fd);      // K=128, N=128
  pk(W_comb, wt_comb, Fd, 2 * Fd);      // K=256, N=128
  pk(W_ih,   wt_ih,   4 * Hd, 4 * Fd);  // K=512, N=2048
  pk(W_hh,   wt_hh,   4 * Hd, Hd);      // K=512, N=2048

  for (int t = 0; t < Tlen; ++t) {
    rits_step<<<Bsz / 16, 256, 0, stream>>>(
        data, wt_dh, b_dh, w_dx, b_dx, wt_hist, b_hist, wt_feat, b_feat, W_feat,
        wt_comb, b_comb, wt_ih, wt_hh, b_ih, b_hh,
        hstate, cstate, lossacc, imps, t);
  }

  rits_fin<<<1, 1, 0, stream>>>(lossacc, out);
}
